// ODConv2d_fusion_49950469653109
// MI455X (gfx1250) — matrix operand order repaired and checked by reading, not compiled, not run on hardware
//
#include <hip/hip_runtime.h>
#include <hip/hip_bf16.h>

// ---------------------------------------------------------------------------
// ODConv2d fusion for MI455X (gfx1250, wave32, WMMA bf16 16x16x32 + TDM).
//
// Algebra: fold channel/filter/spatial/kernel/dif attentions AND the 1x1 fuse
// conv into per-batch effective 3x3 weights Weff[s][b][tap][o][cin] (bf16).
// Then:  out = conv(x0,Weff0) + conv(x1,Weff1) + fuse_b   (f32 accum).
//
// Weight tiles are staged LDS-side by the Tensor Data Mover when the
// tensor_load_to_lds builtin is available (TENSORcnt-tracked), else by a
// synchronous vector copy.
// ---------------------------------------------------------------------------

#define B_    8
#define CIN_  128
#define COUT_ 128
#define KN_   4
#define H_    160
#define W_    160
#define ATT_  16
#define MID_  32
#define TAPS  9
#define HW_   (H_*W_)

typedef __attribute__((ext_vector_type(16))) __bf16 v16bf;
typedef __attribute__((ext_vector_type(8)))  float  v8f;

#if defined(__has_builtin)
#if __has_builtin(__builtin_amdgcn_tensor_load_to_lds) && \
    __has_builtin(__builtin_amdgcn_s_wait_tensorcnt)
#define USE_TDM 1
#endif
#endif

#ifdef USE_TDM
typedef unsigned int tdm_u32x4 __attribute__((ext_vector_type(4)));
typedef int          tdm_i32x4 __attribute__((ext_vector_type(4)));
typedef int          tdm_i32x8 __attribute__((ext_vector_type(8)));
#endif

union FragAB { v16bf v; uint4 q[2]; };

__device__ __forceinline__ unsigned short f2bf(float f) {
  unsigned int u = __float_as_uint(f);
  u += 0x7FFFu + ((u >> 16) & 1u);        // round-to-nearest-even
  return (unsigned short)(u >> 16);
}
__device__ __forceinline__ float sigm(float x) { return 1.f / (1.f + __expf(-x)); }

// workspace layout (float offsets)
#define OF_GAP0  0
#define OF_GAP1  1024
#define OF_FDAVG 2048
#define OF_FDMAX 3072
#define OF_CA    4096            // [2][8][128]
#define OF_FA    6144            // [2][8][128]
#define OF_SAMFD 8192            // [2][8][9]  (sa * Mfd)
#define OF_KA    8336            // [2][8][4]
#define OF_WMIX  8448            // [2][8][9][128][128] f32
#define WMIX_N   (2*8*9*128*128)
#define OF_WEFF_F (OF_WMIX + WMIX_N)   // ushort (bf16) region starts here (byte off *4, 16B aligned)

// ---------------------------------------------------------------------------
// 1) Per-(b,c) spatial reductions: gap(x0), gap(x1), mean(x0-x1), max(x0-x1)
// ---------------------------------------------------------------------------
__global__ void reduce_kernel(const float* __restrict__ x0,
                              const float* __restrict__ x1,
                              float* __restrict__ ws) {
  const int bc = blockIdx.x;                 // b*128 + c
  const float* p0 = x0 + (size_t)bc * HW_;
  const float* p1 = x1 + (size_t)bc * HW_;
  float s0 = 0.f, s1 = 0.f, sd = 0.f, md = -3.4e38f;
  for (int i = threadIdx.x; i < HW_; i += 256) {
    float a = p0[i], b = p1[i];
    s0 += a; s1 += b;
    float d = a - b;
    sd += d; md = fmaxf(md, d);
  }
  __shared__ float r0[256], r1[256], rd[256], rm[256];
  const int t = threadIdx.x;
  r0[t] = s0; r1[t] = s1; rd[t] = sd; rm[t] = md;
  __syncthreads();
  for (int off = 128; off > 0; off >>= 1) {
    if (t < off) {
      r0[t] += r0[t + off]; r1[t] += r1[t + off];
      rd[t] += rd[t + off]; rm[t] = fmaxf(rm[t], rm[t + off]);
    }
    __syncthreads();
  }
  if (t == 0) {
    const float inv = 1.f / (float)HW_;
    ws[OF_GAP0  + bc] = r0[0] * inv;
    ws[OF_GAP1  + bc] = r1[0] * inv;
    ws[OF_FDAVG + bc] = rd[0] * inv;
    ws[OF_FDMAX + bc] = rm[0];
  }
}

// ---------------------------------------------------------------------------
// 2) All tiny attention heads in one block.
// ---------------------------------------------------------------------------
__global__ void attention_kernel(
    const float* __restrict__ fc_w, const float* __restrict__ ln_g, const float* __restrict__ ln_b,
    const float* __restrict__ ch_w, const float* __restrict__ ch_b,
    const float* __restrict__ fl_w, const float* __restrict__ fl_b,
    const float* __restrict__ sp_w, const float* __restrict__ sp_b,
    const float* __restrict__ kn_w, const float* __restrict__ kn_b,
    const float* __restrict__ mlp_w1, const float* __restrict__ mlp_b1,
    const float* __restrict__ mlp_w2, const float* __restrict__ mlp_b2,
    float* __restrict__ ws)
{
  const int t = threadIdx.x;
  __shared__ float s_ha[B_][MID_], s_hm[B_][MID_];
  __shared__ float s_mfd[B_][TAPS];
  __shared__ float s_y[B_][ATT_];
  __shared__ float s_mu[B_], s_rs[B_];

  // dif_atten MLP hidden (8*32 == 256 threads exactly)
  {
    const int b = t >> 5, j = t & 31;
    const float* av = ws + OF_FDAVG + b * CIN_;
    const float* mv = ws + OF_FDMAX + b * CIN_;
    const float* wr = mlp_w1 + j * CIN_;
    float ha = mlp_b1[j], hm = mlp_b1[j];
    for (int c = 0; c < CIN_; ++c) { ha += av[c] * wr[c]; hm += mv[c] * wr[c]; }
    s_ha[b][j] = fmaxf(ha, 0.f);
    s_hm[b][j] = fmaxf(hm, 0.f);
  }
  __syncthreads();
  if (t < B_ * TAPS) {
    const int b = t / TAPS, tt = t % TAPS;
    float acc = 2.f * mlp_b2[tt];                 // b2 added in both mlp() calls
    for (int j = 0; j < MID_; ++j) acc += (s_ha[b][j] + s_hm[b][j]) * mlp_w2[tt * MID_ + j];
    s_mfd[b][tt] = sigm(acc);
  }
  __syncthreads();

  for (int s = 0; s < 2; ++s) {
    const float* gap = ws + (s ? OF_GAP1 : OF_GAP0);
    if (t < B_ * ATT_) {
      const int b = t / ATT_, a = t % ATT_;
      const float* wr = fc_w + a * CIN_;
      const float* g  = gap + b * CIN_;
      float acc = 0.f;
      for (int c = 0; c < CIN_; ++c) acc += g[c] * wr[c];
      s_y[b][a] = acc;
    }
    __syncthreads();
    if (t < B_) {
      float mu = 0.f;
      for (int a = 0; a < ATT_; ++a) mu += s_y[t][a];
      mu *= (1.f / ATT_);
      float v = 0.f;
      for (int a = 0; a < ATT_; ++a) { float d = s_y[t][a] - mu; v += d * d; }
      v *= (1.f / ATT_);
      s_mu[t] = mu; s_rs[t] = rsqrtf(v + 1e-5f);
    }
    __syncthreads();
    if (t < B_ * ATT_) {
      const int b = t / ATT_, a = t % ATT_;
      float y = (s_y[b][a] - s_mu[b]) * s_rs[b] * ln_g[a] + ln_b[a];
      s_y[b][a] = fmaxf(y, 0.f);
    }
    __syncthreads();
    // channel + filter attentions (1024 outputs each)
    for (int k = 0; k < 4; ++k) {
      const int idx = t + k * 256;
      const int b = idx >> 7, c = idx & 127;
      float aA = ch_b[c], aB = fl_b[c];
      for (int a = 0; a < ATT_; ++a) {
        const float y = s_y[b][a];
        aA += y * ch_w[c * ATT_ + a];
        aB += y * fl_w[c * ATT_ + a];
      }
      ws[OF_CA + (s * B_ + b) * CIN_  + c] = sigm(aA);
      ws[OF_FA + (s * B_ + b) * COUT_ + c] = sigm(aB);
    }
    if (t < B_ * TAPS) {
      const int b = t / TAPS, tt = t % TAPS;
      float acc = sp_b[tt];
      for (int a = 0; a < ATT_; ++a) acc += s_y[b][a] * sp_w[tt * ATT_ + a];
      ws[OF_SAMFD + (s * B_ + b) * TAPS + tt] = sigm(acc) * s_mfd[b][tt];
    }
    if (t < B_) {
      float lg[KN_]; float mx = -3.4e38f;
      for (int n = 0; n < KN_; ++n) {
        float acc = kn_b[n];
        for (int a = 0; a < ATT_; ++a) acc += s_y[t][a] * kn_w[n * ATT_ + a];
        lg[n] = acc; mx = fmaxf(mx, acc);
      }
      float sum = 0.f;
      for (int n = 0; n < KN_; ++n) { lg[n] = __expf(lg[n] - mx); sum += lg[n]; }
      const float inv = 1.f / sum;
      for (int n = 0; n < KN_; ++n) ws[OF_KA + (s * B_ + t) * KN_ + n] = lg[n] * inv;
    }
    __syncthreads();
  }
}

// ---------------------------------------------------------------------------
// 3) Wmix[s][b][t][c][cin] = (sum_n ka*W[n,c,cin,t]) * saMfd[s,b,t] * ca[s,b,cin]
// ---------------------------------------------------------------------------
__global__ void wmix_kernel(const float* __restrict__ weight, float* __restrict__ ws) {
  const int idx = blockIdx.x * 256 + threadIdx.x;      // < WMIX_N
  const int cin = idx & 127;
  const int c   = (idx >> 7) & 127;
  const int v   = idx >> 14;                           // t + 9*(b + 8*s)
  const int t   = v % 9;
  const int bs  = v / 9;
  const int b   = bs & 7;
  const int s   = bs >> 3;
  const float* ka = ws + OF_KA + (s * B_ + b) * KN_;
  float acc = 0.f;
#pragma unroll
  for (int n = 0; n < KN_; ++n)
    acc += ka[n] * weight[(((size_t)n * COUT_ + c) * CIN_ + cin) * TAPS + t];
  acc *= ws[OF_SAMFD + (s * B_ + b) * TAPS + t] * ws[OF_CA + (s * B_ + b) * CIN_ + cin];
  ws[OF_WMIX + idx] = acc;
}

// ---------------------------------------------------------------------------
// 4) Weff[s][b][t][o][cin] (bf16) = sum_c fuse_w[o, c+s*128]*fa[s,b,c]*Wmix[..c,cin]
// ---------------------------------------------------------------------------
__global__ void weff_kernel(const float* __restrict__ fuse_w, float* __restrict__ ws,
                            unsigned short* __restrict__ weff) {
  const int blk = blockIdx.x;        // (s*8+b)*9 + t
  const int sb  = blk / 9;
  const int s   = sb >> 3;
  const int b   = sb & 7;
  __shared__ float s_fa[COUT_];
  if (threadIdx.x < COUT_) s_fa[threadIdx.x] = ws[OF_FA + (s * B_ + b) * COUT_ + threadIdx.x];
  __syncthreads();
  const float* wm = ws + OF_WMIX + (size_t)blk * COUT_ * CIN_;     // [c][cin]
  const int cin = threadIdx.x & 127;
  const int o0  = threadIdx.x >> 7;
  for (int i = 0; i < 64; ++i) {
    const int o = o0 + 2 * i;
    const float* fw = fuse_w + o * (2 * COUT_) + s * COUT_;
    float acc = 0.f;
    for (int c = 0; c < COUT_; ++c) acc += fw[c] * s_fa[c] * wm[c * CIN_ + cin];
    weff[((size_t)blk * COUT_ + o) * CIN_ + cin] = f2bf(acc);
  }
}

// ---------------------------------------------------------------------------
// 5) Main conv: implicit GEMM, bf16 WMMA 16x16x32, f32 accum.
//    Block: 64 o-channels x (2 rows x 32 cols); 8 waves = 2(M) x 4(N).
//    Weight tile (3D: 32 cin x 64 o x 9 taps, 16B pad per 64B row) staged by
//    the Tensor Data Mover when available.
// ---------------------------------------------------------------------------
__global__ void __launch_bounds__(256) conv_kernel(
    const float* __restrict__ x0, const float* __restrict__ x1,
    const unsigned short* __restrict__ weff, const float* __restrict__ fuse_b,
    float* __restrict__ out)
{
  __shared__ unsigned short wlds[TAPS * 64 * 40];   // [tap][o(64)][cin pad40] bf16, 46080 B
  __shared__ unsigned short xlds[4 * 34 * 40];      // [row(4)][col(34)][cin pad40] bf16, 10880 B

  const int tid  = threadIdx.x;
  const int lane = tid & 31, wid = tid >> 5;
  const int wm = wid & 1;                // M group: o base wm*32
  const int wn = wid >> 1;               // N group: spatial base wn*16
  const int b  = blockIdx.z >> 1;
  const int oh = blockIdx.z & 1;
  const int w0 = blockIdx.x * 32;
  const int hb = blockIdx.y * 2;
  const int obase = oh * 64;

  v8f acc[2];
#pragma unroll
  for (int i = 0; i < 2; ++i)
#pragma unroll
    for (int r = 0; r < 8; ++r) acc[i][r] = 0.f;

  for (int s = 0; s < 2; ++s) {
    const float* __restrict__ x = s ? x1 : x0;
    const unsigned short* __restrict__ Wf =
        weff + (size_t)(s * B_ + b) * TAPS * COUT_ * CIN_;

    for (int cc = 0; cc < 4; ++cc) {
      const int cin0 = cc * 32;
      __syncthreads();
      // ---- stage input tile (f32 -> bf16), rows hb-1..hb+2, cols w0-1..w0+32
      for (int e = tid; e < 32 * 136; e += 256) {
        const int ci = e / 136;
        const int rc = e - ci * 136;
        const int r  = rc / 34;
        const int c  = rc - r * 34;
        const int gh = hb + r - 1;
        const int gw = w0 + c - 1;
        float v = 0.f;
        if (gh >= 0 && gh < H_ && gw >= 0 && gw < W_)
          v = x[(((size_t)b * CIN_ + cin0 + ci) * H_ + gh) * W_ + gw];
        xlds[(r * 34 + c) * 40 + ci] = f2bf(v);
      }
      if (cc < 3)     // hint next input chunk toward cache (global_prefetch_b8)
        __builtin_prefetch(&x[(((size_t)b * CIN_ + cin0 + 32) * H_ + hb) * W_ + w0], 0, 1);

#ifdef USE_TDM
      // ---- Tensor Data Mover: 3D tile  dim0=32 cin (64B) -> +16B LDS pad,
      //      dim1=64 o rows @ stride 128 el, dim2=9 taps @ stride 16384 el.
      if (tid < 32) {   // one wave issues the DMA (TDM ignores EXEC)
        const unsigned long long ga =
            (unsigned long long)(uintptr_t)(Wf + (size_t)obase * CIN_ + cin0);
        const unsigned int ldsa =
            (unsigned int)(unsigned long long)(uintptr_t)(&wlds[0]);
        tdm_u32x4 g0;
        g0[0] = 1u;                                   // count=1, user D#
        g0[1] = ldsa;                                 // lds_addr (bytes)
        g0[2] = (unsigned int)ga;                     // global_addr[31:0]
        g0[3] = (unsigned int)((ga >> 32) & 0x01FFFFFFull) | (2u << 30); // addr[56:32] | type=2
        tdm_i32x8 g1;
        g1[0] = (int)((1u << 16)        // data_size = 2 bytes
                    | (1u << 20)        // pad_enable
                    | (3u << 22)        // pad_interval: 64B between pads
                    | (3u << 25));      // pad_amount: 4 DWORDs (16B)
        g1[1] = (int)(128u << 16);      // tensor_dim0 = 128 (low 16 in [31:16])
        g1[2] = (int)(128u << 16);      // tensor_dim0 hi=0 | tensor_dim1 = 128
        g1[3] = (int)(32u << 16);       // tensor_dim1 hi=0 | tile_dim0 = 32
        g1[4] = (int)(64u | (9u << 16));// tile_dim1 = 64 | tile_dim2 = 9
        g1[5] = 128;                    // tensor_dim0_stride = 128 el
        g1[6] = (int)(0x4000u << 16);   // stride0 hi=0 | tensor_dim1_stride lo = 16384
        g1[7] = 0;                      // tensor_dim1_stride hi
        tdm_i32x4 g2;
        g2[0] = 9; g2[1] = 0; g2[2] = 0; g2[3] = 0;   // tensor_dim2 = 9
        tdm_i32x4 g3;
        g3[0] = 0; g3[1] = 0; g3[2] = 0; g3[3] = 0;
#if defined(__clang_major__) && (__clang_major__ >= 23)
        tdm_i32x8 g4;
#pragma unroll
        for (int i = 0; i < 8; ++i) g4[i] = 0;
        __builtin_amdgcn_tensor_load_to_lds(g0, g1, g2, g3, g4, 0);
#else
        __builtin_amdgcn_tensor_load_to_lds(g0, g1, g2, g3, 0);
#endif
        __builtin_amdgcn_s_wait_tensorcnt(0);
      }
#else
      // ---- fallback: synchronous vector copy of the weight tile
      for (int e = tid; e < TAPS * 64 * 4; e += 256) {
        const int t  = e >> 8;
        const int oc = e & 255;
        const int ol = oc >> 2;
        const int g  = oc & 3;
        const uint4* src =
            (const uint4*)&Wf[(size_t)(t * COUT_ + obase + ol) * CIN_ + cin0 + g * 8];
        *(uint4*)&wlds[(t * 64 + ol) * 40 + g * 8] = *src;
      }
      if (cc < 3)
        __builtin_prefetch(&Wf[(size_t)obase * CIN_ + cin0 + 32], 0, 1);
#endif
      __syncthreads();

      // ---- compute: 9 taps x 2 M-frags of v_wmma_f32_16x16x32_bf16
      const int nl  = lane & 15;
      const int sp  = wn * 16 + nl;          // tile-local spatial 0..63
      const int lr0 = sp >> 5;               // local row 0..1
      const int lc0 = sp & 31;               // local col 0..31
      const int ci0 = (lane >> 4) << 3;      // B: K half offset {0,8} (K = ci0..ci0+7, ci0+16..ci0+23)
      const int ml  = lane & 15;             // A: M = lane%16
      const int kh8 = (lane >> 4) << 3;      // A: K half offset {0,8}
#pragma unroll
      for (int t = 0; t < TAPS; ++t) {
        const int kr = t / 3;
        const int kc = t - 3 * kr;
        FragAB fB;
        const uint4* bp = (const uint4*)&xlds[((lr0 + kr) * 34 + (lc0 + kc)) * 40 + ci0];
        fB.q[0] = bp[0];
        fB.q[1] = bp[2];
#pragma unroll
        for (int mi = 0; mi < 2; ++mi) {
          FragAB fA;
          const int ol = wm * 32 + mi * 16 + ml;
          const unsigned short* ap = &wlds[(t * 64 + ol) * 40 + kh8];
          fA.q[0] = *(const uint4*)ap;          // K = kh8..kh8+7
          fA.q[1] = *(const uint4*)(ap + 16);   // K = kh8+16..kh8+23
          acc[mi] = __builtin_amdgcn_wmma_f32_16x16x32_bf16(
              false, fA.v, false, fB.v, (short)0, acc[mi], false, false);
        }
      }
    }
  }

  // ---- store (C layout: VGPR r -> M = r + 8*(lane/16); lane%16 -> N)
  {
    const int nl = lane & 15;
    const int sp = wn * 16 + nl;
    const int gh = hb + (sp >> 5);
    const int gw = w0 + (sp & 31);
    const int mhi = (lane >> 4) << 3;
#pragma unroll
    for (int mi = 0; mi < 2; ++mi) {
#pragma unroll
      for (int r = 0; r < 8; ++r) {
        const int o = obase + wm * 32 + mi * 16 + r + mhi;
        out[(((size_t)b * COUT_ + o) * H_ + gh) * W_ + gw] = acc[mi][r] + fuse_b[o];
      }
    }
  }
}

// ---------------------------------------------------------------------------
extern "C" void kernel_launch(void* const* d_in, const int* in_sizes, int n_in,
                              void* d_out, int out_size, void* d_ws, size_t ws_size,
                              hipStream_t stream) {
  const float* x0     = (const float*)d_in[0];
  const float* x1     = (const float*)d_in[1];
  const float* fc_w   = (const float*)d_in[2];
  const float* ln_g   = (const float*)d_in[3];
  const float* ln_b   = (const float*)d_in[4];
  const float* ch_w   = (const float*)d_in[5];
  const float* ch_b   = (const float*)d_in[6];
  const float* fl_w   = (const float*)d_in[7];
  const float* fl_b   = (const float*)d_in[8];
  const float* sp_w   = (const float*)d_in[9];
  const float* sp_b   = (const float*)d_in[10];
  const float* kn_w   = (const float*)d_in[11];
  const float* kn_b   = (const float*)d_in[12];
  const float* weight = (const float*)d_in[13];
  const float* mlp_w1 = (const float*)d_in[14];
  const float* mlp_b1 = (const float*)d_in[15];
  const float* mlp_w2 = (const float*)d_in[16];
  const float* mlp_b2 = (const float*)d_in[17];
  const float* fuse_w = (const float*)d_in[18];
  const float* fuse_b = (const float*)d_in[19];

  float* ws = (float*)d_ws;
  unsigned short* weff = (unsigned short*)((char*)d_ws + (size_t)OF_WEFF_F * 4);
  float* out = (float*)d_out;

  reduce_kernel<<<B_ * CIN_, 256, 0, stream>>>(x0, x1, ws);
  attention_kernel<<<1, 256, 0, stream>>>(fc_w, ln_g, ln_b, ch_w, ch_b, fl_w, fl_b,
                                          sp_w, sp_b, kn_w, kn_b,
                                          mlp_w1, mlp_b1, mlp_w2, mlp_b2, ws);
  wmix_kernel<<<WMIX_N / 256, 256, 0, stream>>>(weight, ws);
  weff_kernel<<<2 * B_ * TAPS, 256, 0, stream>>>(fuse_w, ws, weff);

  dim3 grid(W_ / 32, H_ / 2, B_ * 2);   // (5, 80, 16)
  conv_kernel<<<grid, 256, 0, stream>>>(x0, x1, weff, fuse_b, out);
}
